// WindowAttention_50233937494384
// MI455X (gfx1250) — matrix operand order repaired and checked by reading, not compiled, not run on hardware
//
#include <hip/hip_runtime.h>

#define NTOK 16384
#define EDIM 1024
#define HDIM 4096

typedef __attribute__((ext_vector_type(16))) __bf16 v16bf;
typedef __attribute__((ext_vector_type(8)))  __bf16 v8bf;
typedef __attribute__((ext_vector_type(8)))  float  v8f;

__device__ __forceinline__ unsigned short f2bf(float f) {
    union { float f; unsigned int u; } c; c.f = f;
    unsigned int u = c.u;
    u += 0x7FFFu + ((u >> 16) & 1u);          // round-to-nearest-even
    return (unsigned short)(u >> 16);
}

__device__ __forceinline__ v16bf cat8(v8bf lo, v8bf hi) {
    return __builtin_shufflevector(lo, hi, 0,1,2,3,4,5,6,7,8,9,10,11,12,13,14,15);
}

// ---------------------------------------------------------------- f32 -> bf16
__global__ __launch_bounds__(256)
void cvt_bf16_kernel(const float* __restrict__ s, unsigned short* __restrict__ d, int n) {
    int i = (blockIdx.x * 256 + threadIdx.x) * 4;
    if (i + 3 < n) {
        float4 f = *(const float4*)(s + i);
        unsigned short o[4] = { f2bf(f.x), f2bf(f.y), f2bf(f.z), f2bf(f.w) };
        *(uint2*)(d + i) = *(const uint2*)o;
    }
}

// ------------------------------------------- f32 RxC -> bf16 CxR (transpose)
__global__ __launch_bounds__(256)
void cvt_bf16_t_kernel(const float* __restrict__ s, unsigned short* __restrict__ d,
                       int R, int C) {
    __shared__ float tile[32][33];
    const int tx = threadIdx.x & 31;
    const int ty = threadIdx.x >> 5;          // 0..7
    const int r0 = blockIdx.y * 32;
    const int c0 = blockIdx.x * 32;
    #pragma unroll
    for (int j = 0; j < 4; ++j)
        tile[ty + j * 8][tx] = s[(size_t)(r0 + ty + j * 8) * C + c0 + tx];
    __syncthreads();
    #pragma unroll
    for (int j = 0; j < 4; ++j)
        d[(size_t)(c0 + ty + j * 8) * R + r0 + tx] = f2bf(tile[tx][ty + j * 8]);
}

// ---------------------------------------------------- generic bf16 WMMA GEMM
// C[M,N] = A[M,K] @ BT[N,K]^T + bias. Block tile 128x128, 8 waves (4x2),
// wave tile 32x64 (8 accumulators), K-step 32 via v_wmma_f32_16x16x32_bf16.
// BT is pre-transposed (N-major), so both LDS stages are contiguous b128.
template<int RELU, int OUT_BF16, int TRANS>
__global__ __launch_bounds__(256)
void gemm_bf16_kernel(const unsigned short* __restrict__ A,
                      const unsigned short* __restrict__ BT,
                      const float* __restrict__ bias,
                      float* __restrict__ Cf, unsigned short* __restrict__ Cb,
                      int N, int K, int ldc) {
    constexpr int LDT = 40;                      // halfs, padded stride
    __shared__ __align__(16) unsigned short sA[128 * LDT];
    __shared__ __align__(16) unsigned short sB[128 * LDT];

    const int tid  = threadIdx.x;
    const int wave = tid >> 5, lane = tid & 31;
    const int wm   = wave & 3, wn = wave >> 2;   // 4 x 2 wave grid
    const int bm   = blockIdx.x * 128;
    const int bn   = blockIdx.y * 128;

    v8f acc[2][4] = {};

    for (int k0 = 0; k0 < K; k0 += 32) {
        __syncthreads();
        // stage A: 128 rows x 32 halfs (16B chunks), 2 per thread
        // stage BT: 128 rows (n) x 32 halfs (k), 2 per thread
        #pragma unroll
        for (int i = 0; i < 2; ++i) {
            int idx = tid + i * 256;             // 0..511
            int r   = idx >> 2;                  // 0..127
            int c8  = (idx & 3) * 8;
            *(uint4*)&sA[r * LDT + c8] =
                *(const uint4*)(A + (size_t)(bm + r) * K + k0 + c8);
            *(uint4*)&sB[r * LDT + c8] =
                *(const uint4*)(BT + (size_t)(bn + r) * K + k0 + c8);
            if (k0 + 32 < K) {
                __builtin_prefetch(A  + (size_t)(bm + r) * K + k0 + 32 + c8, 0, 0);
                __builtin_prefetch(BT + (size_t)(bn + r) * K + k0 + 32 + c8, 0, 0);
            }
        }
        __syncthreads();

        // A fragments (16x32): lane<16 -> K 0..7 & 16..23 of row lane;
        //                      lane>=16 -> K 8..15 & 24..31
        const int lane15 = lane & 15;
        const int ak = (lane < 16) ? 0 : 8;
        const int bk = (lane < 16) ? 0 : 8;
        v16bf afr[2];
        #pragma unroll
        for (int mt = 0; mt < 2; ++mt) {
            int arow = wm * 32 + mt * 16 + lane15;
            afr[mt] = cat8(*(const v8bf*)&sA[arow * LDT + ak],
                           *(const v8bf*)&sA[arow * LDT + ak + 16]);
        }
        #pragma unroll
        for (int t = 0; t < 4; ++t) {
            int bcol = wn * 64 + t * 16 + lane15;
            v16bf bfv = cat8(*(const v8bf*)&sB[bcol * LDT + bk],
                             *(const v8bf*)&sB[bcol * LDT + bk + 16]);
            #pragma unroll
            for (int mt = 0; mt < 2; ++mt)
                acc[mt][t] = __builtin_amdgcn_wmma_f32_16x16x32_bf16(
                                 false, afr[mt], false, bfv, (short)0,
                                 acc[mt][t], false, false);
        }
    }

    // epilogue: C/D layout — VGPR j: lanes 0-15 M=j, lanes 16-31 M=j+8; N=lane%16
    const int col = lane & 15;
    #pragma unroll
    for (int mt = 0; mt < 2; ++mt) {
        int r0 = bm + wm * 32 + mt * 16 + (lane >> 4) * 8;
        #pragma unroll
        for (int t = 0; t < 4; ++t) {
            int c  = bn + wn * 64 + t * 16 + col;
            float bv = bias[c];
            #pragma unroll
            for (int j = 0; j < 8; ++j) {
                float val = acc[mt][t][j] + bv;
                if (RELU) val = fmaxf(val, 0.f);
                size_t off = TRANS ? ((size_t)c * ldc + (r0 + j))
                                   : ((size_t)(r0 + j) * ldc + c);
                if (OUT_BF16) Cb[off] = f2bf(val);
                else          Cf[off] = val;
            }
        }
    }
}

// ------------------------------------------------------- per-window attention
// One block per 64-token window. scores = QK^T/32 (K=1024), softmax, out = P@V.
// V is supplied transposed (E x NTOK) so all staging is contiguous.
__global__ __launch_bounds__(256)
void attn_kernel(const unsigned short* __restrict__ q,
                 const unsigned short* __restrict__ k,
                 const unsigned short* __restrict__ vT,
                 float* __restrict__ out) {
    constexpr int LDT = 72;
    __shared__ __align__(16) unsigned short sQ[64 * LDT];
    __shared__ __align__(16) unsigned short sK[64 * LDT];
    __shared__ __align__(16) float          sS[64 * 64];
    __shared__ __align__(16) unsigned short sP[64 * LDT];
    __shared__ __align__(16) unsigned short sV[128 * LDT];

    const int tid  = threadIdx.x;
    const int wave = tid >> 5, lane = tid & 31;
    const int wm   = wave >> 1, wn = wave & 1;
    const size_t t0 = (size_t)blockIdx.x * 64;   // window token base

    // ---------- phase 1: scores (64x64, K = 1024 in chunks of 64) ----------
    v8f sacc[2] = {};
    for (int kc = 0; kc < EDIM; kc += 64) {
        __syncthreads();
        #pragma unroll
        for (int i = 0; i < 2; ++i) {
            int idx = tid + i * 256;             // 0..511
            int r   = idx >> 3;                  // 0..63
            int c8  = (idx & 7) * 8;
            *(uint4*)&sQ[r * LDT + c8] = *(const uint4*)(q + (t0 + r) * EDIM + kc + c8);
            *(uint4*)&sK[r * LDT + c8] = *(const uint4*)(k + (t0 + r) * EDIM + kc + c8);
        }
        __syncthreads();
        #pragma unroll
        for (int ks = 0; ks < 64; ks += 32) {
            int arow = wm * 16 + (lane & 15);
            int ak   = ks + ((lane < 16) ? 0 : 8);
            v16bf af = cat8(*(const v8bf*)&sQ[arow * LDT + ak],
                            *(const v8bf*)&sQ[arow * LDT + ak + 16]);
            #pragma unroll
            for (int t = 0; t < 2; ++t) {
                int bcol = wn * 32 + t * 16 + (lane & 15);
                int bk   = ks + ((lane < 16) ? 0 : 8);
                v16bf bfv = cat8(*(const v8bf*)&sK[bcol * LDT + bk],
                                 *(const v8bf*)&sK[bcol * LDT + bk + 16]);
                sacc[t] = __builtin_amdgcn_wmma_f32_16x16x32_bf16(
                              false, af, false, bfv, (short)0, sacc[t], false, false);
            }
        }
    }
    {   // spill scores to LDS, scaled by E^-0.5 = 1/32
        int r0 = wm * 16 + (lane >> 4) * 8;
        int c  = lane & 15;
        #pragma unroll
        for (int t = 0; t < 2; ++t) {
            int cc = wn * 32 + t * 16 + c;
            #pragma unroll
            for (int j = 0; j < 8; ++j)
                sS[(r0 + j) * 64 + cc] = sacc[t][j] * 0.03125f;
        }
    }
    __syncthreads();

    // ---------- phase 2: row softmax -> bf16 P ----------
    if (tid < 64) {
        float mx = -1e30f;
        for (int c = 0; c < 64; ++c) mx = fmaxf(mx, sS[tid * 64 + c]);
        float sum = 0.f;
        for (int c = 0; c < 64; ++c) {
            float e = __expf(sS[tid * 64 + c] - mx);
            sS[tid * 64 + c] = e; sum += e;
        }
        float inv = 1.f / sum;
        for (int c = 0; c < 64; ++c) sP[tid * LDT + c] = f2bf(sS[tid * 64 + c] * inv);
    }

    // ---------- phase 3: out = P(64x64) @ V(64x1024), N chunks of 128 ----------
    for (int nc = 0; nc < EDIM; nc += 128) {
        __syncthreads();
        #pragma unroll
        for (int i = 0; i < 4; ++i) {            // stage V^T rows: sV[n][k], contiguous
            int idx = tid + i * 256;             // 0..1023
            int r   = idx >> 3;                  // 0..127
            int c8  = (idx & 7) * 8;
            *(uint4*)&sV[r * LDT + c8] =
                *(const uint4*)(vT + (size_t)(nc + r) * NTOK + t0 + c8);
        }
        __syncthreads();
        v8f oacc[4] = {};
        #pragma unroll
        for (int ks = 0; ks < 64; ks += 32) {
            int arow = wm * 16 + (lane & 15);
            int ak   = ks + ((lane < 16) ? 0 : 8);
            v16bf af = cat8(*(const v8bf*)&sP[arow * LDT + ak],
                            *(const v8bf*)&sP[arow * LDT + ak + 16]);
            #pragma unroll
            for (int t = 0; t < 4; ++t) {
                int bcol = wn * 64 + t * 16 + (lane & 15);
                int bk   = ks + ((lane < 16) ? 0 : 8);
                v16bf bfv = cat8(*(const v8bf*)&sV[bcol * LDT + bk],
                                 *(const v8bf*)&sV[bcol * LDT + bk + 16]);
                oacc[t] = __builtin_amdgcn_wmma_f32_16x16x32_bf16(
                              false, af, false, bfv, (short)0, oacc[t], false, false);
            }
        }
        int r0 = wm * 16 + (lane >> 4) * 8;
        int c  = lane & 15;
        #pragma unroll
        for (int t = 0; t < 4; ++t) {
            int cc = nc + wn * 64 + t * 16 + c;
            #pragma unroll
            for (int j = 0; j < 8; ++j)
                out[(t0 + r0 + j) * EDIM + cc] = oacc[t][j];
        }
    }
}

// ---------------------------------------------- residual + LayerNorm (row=1024)
template<int EMIT_BF16>
__global__ __launch_bounds__(256)
void resln_kernel(const float* __restrict__ x, const float* __restrict__ r,
                  const float* __restrict__ g, const float* __restrict__ be,
                  float* __restrict__ y, unsigned short* __restrict__ yb) {
    __shared__ float sred[8];
    const size_t row = blockIdx.x;
    const int tid = threadIdx.x, lane = tid & 31, wave = tid >> 5;

    float vals[4];
    float s = 0.f;
    #pragma unroll
    for (int i = 0; i < 4; ++i) {
        int c = tid + i * 256;
        float t = x[row * EDIM + c] + r[row * EDIM + c];
        vals[i] = t; s += t;
    }
    #pragma unroll
    for (int o = 16; o > 0; o >>= 1) s += __shfl_xor(s, o, 32);
    if (lane == 0) sred[wave] = s;
    __syncthreads();
    float tot = 0.f;
    #pragma unroll
    for (int i = 0; i < 8; ++i) tot += sred[i];
    float mean = tot * (1.f / EDIM);
    __syncthreads();

    float vs = 0.f;
    #pragma unroll
    for (int i = 0; i < 4; ++i) { float d = vals[i] - mean; vs += d * d; }
    #pragma unroll
    for (int o = 16; o > 0; o >>= 1) vs += __shfl_xor(vs, o, 32);
    if (lane == 0) sred[wave] = vs;
    __syncthreads();
    float vtot = 0.f;
    #pragma unroll
    for (int i = 0; i < 8; ++i) vtot += sred[i];
    float rstd = rsqrtf(vtot * (1.f / EDIM) + 1e-5f);

    #pragma unroll
    for (int i = 0; i < 4; ++i) {
        int c = tid + i * 256;
        float o = (vals[i] - mean) * rstd * g[c] + be[c];
        y[row * EDIM + c] = o;
        if (EMIT_BF16) yb[row * EDIM + c] = f2bf(o);
    }
}

// ----------------------------------------------------------------- launcher
extern "C" void kernel_launch(void* const* d_in, const int* in_sizes, int n_in,
                              void* d_out, int out_size, void* d_ws, size_t ws_size,
                              hipStream_t stream) {
    const float* x   = (const float*)d_in[0];
    const float* Wq  = (const float*)d_in[1];
    const float* bq  = (const float*)d_in[2];
    const float* Wk  = (const float*)d_in[3];
    const float* bk  = (const float*)d_in[4];
    const float* Wv  = (const float*)d_in[5];
    const float* bv  = (const float*)d_in[6];
    const float* g1  = (const float*)d_in[7];
    const float* be1 = (const float*)d_in[8];
    const float* W1  = (const float*)d_in[9];
    const float* b1  = (const float*)d_in[10];
    const float* W2  = (const float*)d_in[11];
    const float* b2  = (const float*)d_in[12];
    const float* g2  = (const float*)d_in[13];
    const float* be2 = (const float*)d_in[14];

    char* ws = (char*)d_ws;
    size_t off = 0;
    auto carve = [&](size_t bytes) -> void* {
        void* p = ws + off;
        off += (bytes + 255) & ~(size_t)255;
        return p;
    };
    unsigned short* xb  = (unsigned short*)carve((size_t)NTOK * EDIM * 2);
    unsigned short* wqT = (unsigned short*)carve((size_t)EDIM * EDIM * 2);   // N x K
    unsigned short* wkT = (unsigned short*)carve((size_t)EDIM * EDIM * 2);
    unsigned short* wvT = (unsigned short*)carve((size_t)EDIM * EDIM * 2);
    unsigned short* w1T = (unsigned short*)carve((size_t)EDIM * HDIM * 2);   // HID x E
    unsigned short* w2T = (unsigned short*)carve((size_t)HDIM * EDIM * 2);   // E x HID
    unsigned short* qb  = (unsigned short*)carve((size_t)NTOK * EDIM * 2);
    unsigned short* kb  = (unsigned short*)carve((size_t)NTOK * EDIM * 2);
    unsigned short* vbT = (unsigned short*)carve((size_t)NTOK * EDIM * 2);   // E x NTOK
    float*          att = (float*)carve((size_t)NTOK * EDIM * 4);
    float*          yf  = (float*)carve((size_t)NTOK * EDIM * 4);
    unsigned short* ybf = (unsigned short*)carve((size_t)NTOK * EDIM * 2);
    unsigned short* hb  = (unsigned short*)carve((size_t)NTOK * HDIM * 2);
    float*          ff  = att;   // reuse: attention-out consumed before FFN2 writes

    // x -> bf16; weights -> bf16 transposed (N-major)
    cvt_bf16_kernel<<<(NTOK * EDIM) / 1024, 256, 0, stream>>>(x, xb, NTOK * EDIM);
    cvt_bf16_t_kernel<<<dim3(EDIM / 32, EDIM / 32), 256, 0, stream>>>(Wq, wqT, EDIM, EDIM);
    cvt_bf16_t_kernel<<<dim3(EDIM / 32, EDIM / 32), 256, 0, stream>>>(Wk, wkT, EDIM, EDIM);
    cvt_bf16_t_kernel<<<dim3(EDIM / 32, EDIM / 32), 256, 0, stream>>>(Wv, wvT, EDIM, EDIM);
    cvt_bf16_t_kernel<<<dim3(HDIM / 32, EDIM / 32), 256, 0, stream>>>(W1, w1T, EDIM, HDIM);
    cvt_bf16_t_kernel<<<dim3(EDIM / 32, HDIM / 32), 256, 0, stream>>>(W2, w2T, HDIM, EDIM);

    // QKV projections (out bf16; V emitted transposed E x NTOK)
    dim3 gQKV(NTOK / 128, EDIM / 128);
    gemm_bf16_kernel<0, 1, 0><<<gQKV, 256, 0, stream>>>(xb, wqT, bq, nullptr, qb,  EDIM, EDIM, EDIM);
    gemm_bf16_kernel<0, 1, 0><<<gQKV, 256, 0, stream>>>(xb, wkT, bk, nullptr, kb,  EDIM, EDIM, EDIM);
    gemm_bf16_kernel<0, 1, 1><<<gQKV, 256, 0, stream>>>(xb, wvT, bv, nullptr, vbT, EDIM, EDIM, NTOK);

    // windowed attention (256 windows of 64 tokens)
    attn_kernel<<<NTOK / 64, 256, 0, stream>>>(qb, kb, vbT, att);

    // residual + LN1 -> y (f32) and y (bf16)
    resln_kernel<1><<<NTOK, 256, 0, stream>>>(x, att, g1, be1, yf, ybf);

    // FFN1: relu(y @ W1 + b1) -> bf16
    dim3 gF1(NTOK / 128, HDIM / 128);
    gemm_bf16_kernel<1, 1, 0><<<gF1, 256, 0, stream>>>(ybf, w1T, b1, nullptr, hb, HDIM, EDIM, HDIM);

    // FFN2: h @ W2 + b2 -> f32
    dim3 gF2(NTOK / 128, EDIM / 128);
    gemm_bf16_kernel<0, 0, 0><<<gF2, 256, 0, stream>>>(hb, w2T, b2, ff, nullptr, EDIM, HDIM, EDIM);

    // residual + LN2 -> final output
    resln_kernel<0><<<NTOK, 256, 0, stream>>>(yf, ff, g2, be2, (float*)d_out, nullptr);
}
